// QattenMixer_New_10977936408644
// MI455X (gfx1250) — matrix operand order repaired and checked by reading, not compiled, not run on hardware
//
#include <hip/hip_runtime.h>
#include <cstdint>
#include <cstddef>

typedef __attribute__((ext_vector_type(16))) __bf16 bf16x16;
typedef __attribute__((ext_vector_type(8)))  float  f32x8;

#define B_TOT 4096
#define SDIM  1200
#define SP    1216          // padded K (38 * 32)
#define N1    2560          // 2048 sel-h1 + 256 wh + 256 v
#define NHEAD 8
#define KHID  128
#define HYP   256

// ---- workspace layout (bytes) ----
constexpr size_t OFF_SBF  = 0;                                   // bf16 [4096][1216]
constexpr size_t OFF_WCAT = OFF_SBF  + (size_t)B_TOT * SP * 2;   // bf16 [2560][1216]
constexpr size_t OFF_BCAT = OFF_WCAT + (size_t)N1 * SP * 2;      // f32  [2560]
constexpr size_t OFF_Y1   = OFF_BCAT + (size_t)N1 * 4;           // bf16 [4096][2560]
constexpr size_t OFF_KW1  = OFF_Y1   + (size_t)B_TOT * N1 * 2;   // bf16 [8][128][32]
constexpr size_t OFF_WQ   = OFF_KW1  + (size_t)NHEAD * KHID * 32 * 2; // f32 [8][128]
constexpr size_t OFF_WM   = OFF_WQ   + (size_t)NHEAD * KHID * 4; // bf16 [8][128][256]
constexpr size_t OFF_SELK = OFF_WM   + (size_t)NHEAD * KHID * HYP * 2; // f32 [4096][8][128]
constexpr size_t OFF_HQ   = OFF_SELK + (size_t)B_TOT * NHEAD * KHID * 4; // f32 [4096][8]
constexpr size_t OFF_WH   = OFF_HQ   + (size_t)B_TOT * NHEAD * 4;        // f32 [4096][8]
constexpr size_t OFF_V    = OFF_WH   + (size_t)B_TOT * NHEAD * 4;        // f32 [4096]
constexpr size_t OFF_ACC  = OFF_V    + (size_t)B_TOT * 4;                // f32 [16]: reg[8], ent[8]

// ---- WMMA helper (CDNA5 16x16x32 bf16, fp32 accum) ----
__device__ __forceinline__ f32x8 wmma_bf16(bf16x16 a, bf16x16 b, f32x8 c) {
  return __builtin_amdgcn_wmma_f32_16x16x32_bf16(false, a, false, b, (short)0, c, false, false);
}

// ---- async global->LDS copy (CDNA5, tracked by ASYNCcnt) ----
__device__ __forceinline__ void async_load_b128(uint32_t lds_off, const void* gaddr) {
  asm volatile("global_load_async_to_lds_b128 %0, %1, off"
               :: "v"(lds_off), "v"(gaddr) : "memory");
}
__device__ __forceinline__ void wait_asynccnt0() {
  asm volatile("s_wait_asynccnt 0x0" ::: "memory");
}

// A fragment 16x32 (row-major source, rows = M): lanes 0-15 K{0..7,16..23}, lanes 16-31 K{8..15,24..31}
__device__ __forceinline__ bf16x16 frag_A16x32(const __bf16* base, int stride_bf, int lane) {
  int r  = lane & 15;
  int kb = (lane < 16) ? 0 : 16;                 // byte offset of first dword group
  const char* p = (const char*)(base + (size_t)r * stride_bf);
  union { uint4 u[2]; bf16x16 v; } f;
  f.u[0] = *(const uint4*)(p + kb);
  f.u[1] = *(const uint4*)(p + kb + 32);
  return f.v;
}

// B fragment 32x16 from row-major weight [n][k]: lanes 0-15 K{0..7,16..23}, lanes 16-31 K{8..15,24..31}
__device__ __forceinline__ bf16x16 frag_B32x16(const __bf16* base, int stride_bf, int lane) {
  int n  = lane & 15;
  int kb = (lane < 16) ? 0 : 16;                 // byte offset
  const char* p = (const char*)(base + (size_t)n * stride_bf);
  union { uint4 u[2]; bf16x16 v; } f;
  f.u[0] = *(const uint4*)(p + kb);
  f.u[1] = *(const uint4*)(p + kb + 32);
  return f.v;
}

// ================= conversion / prep kernels =================
__global__ void k_conv_states(const float* __restrict__ s, __bf16* __restrict__ sbf) {
  size_t idx = (size_t)blockIdx.x * 256 + threadIdx.x;
  if (idx >= (size_t)B_TOT * SP) return;
  int col = (int)(idx % SP);
  size_t row = idx / SP;
  float v = (col < SDIM) ? s[row * SDIM + col] : 0.f;
  sbf[idx] = (__bf16)v;
}

__global__ void k_conv_w(const float* __restrict__ selW1, const float* __restrict__ whW1,
                         const float* __restrict__ vW1, const float* __restrict__ selb1,
                         const float* __restrict__ whb1, const float* __restrict__ vb1,
                         __bf16* __restrict__ wcat, float* __restrict__ bcat) {
  size_t idx = (size_t)blockIdx.x * 256 + threadIdx.x;
  if (idx >= (size_t)N1 * SP) return;
  int col = (int)(idx % SP);
  int row = (int)(idx / SP);
  float v = 0.f;
  if (col < SDIM) {
    if (row < 2048)      v = selW1[(size_t)row * SDIM + col];
    else if (row < 2304) v = whW1[(size_t)(row - 2048) * SDIM + col];
    else                 v = vW1[(size_t)(row - 2304) * SDIM + col];
  }
  wcat[idx] = (__bf16)v;
  if (idx < N1) {
    int r = (int)idx;
    bcat[r] = (r < 2048) ? selb1[r] : (r < 2304 ? whb1[r - 2048] : vb1[r - 2304]);
  }
}

__global__ void k_conv_key(const float* __restrict__ keyW1, __bf16* __restrict__ kw1bf,
                           float* __restrict__ wq) {
  int idx = blockIdx.x * 256 + threadIdx.x;      // over 8*128*33
  if (idx >= NHEAD * KHID * 33) return;
  int hn = idx / 33, u = idx % 33;
  float v = keyW1[idx];
  if (u < 32) kw1bf[hn * 32 + u] = (__bf16)v;
  else        wq[hn] = v;
}

// M[h][k][e] = sum_d sel_W2[h,d,e] * key_W2[h,d,k]
__global__ void k_combine(const float* __restrict__ selW2, const float* __restrict__ keyW2,
                          __bf16* __restrict__ Wm) {
  int idx = blockIdx.x * 256 + threadIdx.x;      // 8*128*256
  if (idx >= NHEAD * KHID * HYP) return;
  int h = idx >> 15, rem = idx & 32767;
  int k = rem >> 8, e = rem & 255;
  float acc = 0.f;
  #pragma unroll 4
  for (int d = 0; d < 64; ++d)
    acc += selW2[((size_t)(h * 64 + d)) * HYP + e] * keyW2[((size_t)(h * 64 + d)) * KHID + k];
  Wm[((size_t)(h * KHID + k)) * HYP + e] = (__bf16)acc;
}

__global__ void k_zero(float* __restrict__ acc) {
  if (threadIdx.x < 16) acc[threadIdx.x] = 0.f;
}

// ======== K1: Y1 = relu(sbf @ Wcat^T + bcat), async double-buffered ========
__global__ __launch_bounds__(256) void k_gemm1(const __bf16* __restrict__ A,
                                               const __bf16* __restrict__ W,
                                               const float* __restrict__ bias,
                                               __bf16* __restrict__ Y) {
  __shared__ alignas(16) __bf16 As[2][128 * 40];
  __shared__ alignas(16) __bf16 Bs[2][128 * 40];
  const int m0 = blockIdx.y * 128, n0 = blockIdx.x * 128;
  const int tid = threadIdx.x, lane = tid & 31;
  const int wm = ((tid >> 5) & 1) * 64, wn = ((tid >> 5) >> 1) * 32;
  const int r0 = tid >> 2, r1 = (tid + 256) >> 2;
  const int q0 = (tid & 3) * 8, q1 = ((tid + 256) & 3) * 8;

  auto issue = [&](int buf, int kc) {
    async_load_b128((uint32_t)(uintptr_t)&As[buf][r0 * 40 + q0],
                    &A[(size_t)(m0 + r0) * SP + kc * 32 + q0]);
    async_load_b128((uint32_t)(uintptr_t)&Bs[buf][r0 * 40 + q0],
                    &W[(size_t)(n0 + r0) * SP + kc * 32 + q0]);
    async_load_b128((uint32_t)(uintptr_t)&As[buf][r1 * 40 + q1],
                    &A[(size_t)(m0 + r1) * SP + kc * 32 + q1]);
    async_load_b128((uint32_t)(uintptr_t)&Bs[buf][r1 * 40 + q1],
                    &W[(size_t)(n0 + r1) * SP + kc * 32 + q1]);
  };

  f32x8 acc[4][2] = {};
  issue(0, 0);
  for (int kc = 0; kc < 38; ++kc) {
    const int cur = kc & 1;
    wait_asynccnt0();        // own async writes into `cur` complete
    __syncthreads();         // all writes of `cur` visible; all reads of `cur^1` finished
    if (kc + 1 < 38) issue(cur ^ 1, kc + 1);   // refills the buffer read at kc-1

    bf16x16 af[4], bw[2];
    #pragma unroll
    for (int mt = 0; mt < 4; ++mt) af[mt] = frag_A16x32(&As[cur][(wm + mt * 16) * 40], 40, lane);
    #pragma unroll
    for (int nt = 0; nt < 2; ++nt) bw[nt] = frag_B32x16(&Bs[cur][(wn + nt * 16) * 40], 40, lane);
    #pragma unroll
    for (int mt = 0; mt < 4; ++mt)
      #pragma unroll
      for (int nt = 0; nt < 2; ++nt)
        acc[mt][nt] = wmma_bf16(af[mt], bw[nt], acc[mt][nt]);
  }
  const int rbase = (lane >= 16) ? 8 : 0, ncol = lane & 15;
  #pragma unroll
  for (int mt = 0; mt < 4; ++mt)
    #pragma unroll
    for (int nt = 0; nt < 2; ++nt) {
      int gn = n0 + wn + nt * 16 + ncol;
      int gm = m0 + wm + mt * 16 + rbase;
      float bv = bias[gn];
      #pragma unroll
      for (int i = 0; i < 8; ++i) {
        float v = fmaxf(acc[mt][nt][i] + bv, 0.f);
        Y[(size_t)(gm + i) * N1 + gn] = (__bf16)v;
      }
    }
}

// ===== K2: selk[b,h,:] = h1[b,h,:] @ Wm[h]^T, async double-buffered =====
__global__ __launch_bounds__(256) void k_selk(const __bf16* __restrict__ Y1,
                                              const __bf16* __restrict__ Wm,
                                              float* __restrict__ selk) {
  __shared__ alignas(16) __bf16 As[2][128 * 40];
  __shared__ alignas(16) __bf16 Bs[2][128 * 40];
  const int h = blockIdx.x, m0 = blockIdx.y * 128;
  const int tid = threadIdx.x, lane = tid & 31;
  const int wm = ((tid >> 5) & 1) * 64, wn = ((tid >> 5) >> 1) * 32;
  const int r0 = tid >> 2, r1 = (tid + 256) >> 2;
  const int q0 = (tid & 3) * 8, q1 = ((tid + 256) & 3) * 8;

  auto issue = [&](int buf, int kc) {
    async_load_b128((uint32_t)(uintptr_t)&As[buf][r0 * 40 + q0],
                    &Y1[(size_t)(m0 + r0) * N1 + h * HYP + kc * 32 + q0]);
    async_load_b128((uint32_t)(uintptr_t)&Bs[buf][r0 * 40 + q0],
                    &Wm[((size_t)h * KHID + r0) * HYP + kc * 32 + q0]);
    async_load_b128((uint32_t)(uintptr_t)&As[buf][r1 * 40 + q1],
                    &Y1[(size_t)(m0 + r1) * N1 + h * HYP + kc * 32 + q1]);
    async_load_b128((uint32_t)(uintptr_t)&Bs[buf][r1 * 40 + q1],
                    &Wm[((size_t)h * KHID + r1) * HYP + kc * 32 + q1]);
  };

  f32x8 acc[4][2] = {};
  issue(0, 0);
  for (int kc = 0; kc < 8; ++kc) {
    const int cur = kc & 1;
    wait_asynccnt0();
    __syncthreads();
    if (kc + 1 < 8) issue(cur ^ 1, kc + 1);

    bf16x16 af[4], bw[2];
    #pragma unroll
    for (int mt = 0; mt < 4; ++mt) af[mt] = frag_A16x32(&As[cur][(wm + mt * 16) * 40], 40, lane);
    #pragma unroll
    for (int nt = 0; nt < 2; ++nt) bw[nt] = frag_B32x16(&Bs[cur][(wn + nt * 16) * 40], 40, lane);
    #pragma unroll
    for (int mt = 0; mt < 4; ++mt)
      #pragma unroll
      for (int nt = 0; nt < 2; ++nt)
        acc[mt][nt] = wmma_bf16(af[mt], bw[nt], acc[mt][nt]);
  }
  const int rbase = (lane >= 16) ? 8 : 0, ncol = lane & 15;
  #pragma unroll
  for (int mt = 0; mt < 4; ++mt)
    #pragma unroll
    for (int nt = 0; nt < 2; ++nt) {
      int n = wn + nt * 16 + ncol;
      int gm = m0 + wm + mt * 16 + rbase;
      #pragma unroll
      for (int i = 0; i < 8; ++i)
        selk[((size_t)(gm + i) * NHEAD + h) * KHID + n] = acc[mt][nt][i];
    }
}

// ============ K3: fused keys/logits/softmax/head_q/entropy/reg ============
__global__ __launch_bounds__(256) void k_logits(const __bf16* __restrict__ sbf,
                                                const float* __restrict__ qv,
                                                const int* __restrict__ actions,
                                                const __bf16* __restrict__ kw1,
                                                const float* __restrict__ wq,
                                                const float* __restrict__ kb1,
                                                const float* __restrict__ selk,
                                                float* __restrict__ hq,
                                                float* __restrict__ accum) {
  __shared__ alignas(16) __bf16 Us[32 * 40];
  __shared__ float Qs[32];
  __shared__ float Ls[NHEAD][32];
  const int b = blockIdx.x;
  const int tid = threadIdx.x, lane = tid & 31, h = tid >> 5;
  if (tid < 128) {
    int a = tid >> 2, q4 = (tid & 3) * 8;
    *(uint4*)&Us[a * 40 + q4] = *(const uint4*)&sbf[(size_t)b * SP + a * 32 + q4];
  }
  if (tid < 32) Qs[tid] = qv[b * 32 + tid];
  __syncthreads();

  bf16x16 af0 = frag_A16x32(&Us[0], 40, lane);
  bf16x16 af1 = frag_A16x32(&Us[16 * 40], 40, lane);
  const int rbase = (lane >= 16) ? 8 : 0, ncol = lane & 15;
  float part0[8] = {}, part1[8] = {};
  #pragma unroll
  for (int nt = 0; nt < 8; ++nt) {
    bf16x16 bw = frag_B32x16(&kw1[((size_t)h * KHID + nt * 16) * 32], 32, lane);
    f32x8 z = {};
    f32x8 d0 = wmma_bf16(af0, bw, z);
    f32x8 d1 = wmma_bf16(af1, bw, z);
    int n = h * KHID + nt * 16 + ncol;
    float wqv = wq[n], bbv = kb1[n];
    float sk = selk[((size_t)b * NHEAD + h) * KHID + nt * 16 + ncol];
    #pragma unroll
    for (int i = 0; i < 8; ++i) {
      float v0 = fmaxf(d0[i] + wqv * Qs[rbase + i] + bbv, 0.f);
      float v1 = fmaxf(d1[i] + wqv * Qs[16 + rbase + i] + bbv, 0.f);
      part0[i] += v0 * sk;
      part1[i] += v1 * sk;
    }
  }
  // reduce over the 16 lanes of each half (columns n%16)
  #pragma unroll
  for (int i = 0; i < 8; ++i) {
    float p0 = part0[i], p1 = part1[i];
    #pragma unroll
    for (int o = 1; o <= 8; o <<= 1) {
      p0 += __shfl_xor(p0, o, 32);
      p1 += __shfl_xor(p1, o, 32);
    }
    part0[i] = p0; part1[i] = p1;
  }
  if (ncol == 0) {
    #pragma unroll
    for (int i = 0; i < 8; ++i) {
      Ls[h][rbase + i]      = part0[i];
      Ls[h][16 + rbase + i] = part1[i];
    }
  }
  __syncthreads();

  // softmax over agents (lane == agent)
  int a = lane;
  float lg = Ls[h][a];
  float sq = lg * lg;                                          // raw logits^2 (regularizer)
  float scaled = (actions[b * 32 + a] == 0) ? -99999999.0f : lg * 0.125f;
  float mx = scaled;
  #pragma unroll
  for (int o = 16; o; o >>= 1) mx = fmaxf(mx, __shfl_xor(mx, o, 32));
  float e = __expf(scaled - mx);
  float s = e;
  #pragma unroll
  for (int o = 16; o; o >>= 1) s += __shfl_xor(s, o, 32);
  float w = e / s;
  float hp = Qs[a] * w;
  float ep = -w * __logf(w + 1e-8f);
  #pragma unroll
  for (int o = 16; o; o >>= 1) {
    hp += __shfl_xor(hp, o, 32);
    ep += __shfl_xor(ep, o, 32);
    sq += __shfl_xor(sq, o, 32);
  }
  if (lane == 0) {
    hq[b * NHEAD + h] = hp;
    atomicAdd(&accum[h], sq);          // reg
    atomicAdd(&accum[8 + h], ep);      // entropy
  }
}

// ================= K4: wh + v second layers =================
__global__ __launch_bounds__(256) void k_whv(const __bf16* __restrict__ Y1,
                                             const float* __restrict__ whW2,
                                             const float* __restrict__ whb2,
                                             const float* __restrict__ vW2,
                                             const float* __restrict__ vb2,
                                             float* __restrict__ whb, float* __restrict__ vb) {
  const int tid = threadIdx.x, lane = tid & 31;
  const int b = blockIdx.x * 8 + (tid >> 5);
  float part[8] = {}; float vp = 0.f;
  for (int jj = 0; jj < 8; ++jj) {
    int j = jj * 32 + lane;
    float y1 = (float)Y1[(size_t)b * N1 + 2048 + j];
    float y2 = (float)Y1[(size_t)b * N1 + 2304 + j];
    #pragma unroll
    for (int h = 0; h < 8; ++h) part[h] += y1 * whW2[h * 256 + j];
    vp += y2 * vW2[j];
  }
  #pragma unroll
  for (int o = 16; o; o >>= 1) {
    #pragma unroll
    for (int h = 0; h < 8; ++h) part[h] += __shfl_xor(part[h], o, 32);
    vp += __shfl_xor(vp, o, 32);
  }
  if (lane < 8) whb[b * 8 + lane] = fabsf(part[lane] + whb2[lane]);
  if (lane == 0) vb[b] = vp + vb2[0];
}

// ================= K5: final combine + scalars =================
__global__ void k_final(const float* __restrict__ whb, const float* __restrict__ hq,
                        const float* __restrict__ vb, const float* __restrict__ accum,
                        float* __restrict__ out) {
  int t = blockIdx.x * 256 + threadIdx.x;
  if (t < B_TOT) {
    float y = vb[t];
    #pragma unroll
    for (int h = 0; h < 8; ++h) y += whb[t * 8 + h] * hq[t * 8 + h];
    out[t] = y;
  } else if (t == B_TOT) {
    float r = 0.f;
    for (int h = 0; h < 8; ++h) r += accum[h];
    out[B_TOT] = 0.001f * r / ((float)B_TOT * 32.f);
  } else if (t < B_TOT + 9) {
    out[t] = accum[8 + (t - B_TOT - 1)] / (float)B_TOT;
  }
}

extern "C" void kernel_launch(void* const* d_in, const int* in_sizes, int n_in,
                              void* d_out, int out_size, void* d_ws, size_t ws_size,
                              hipStream_t stream) {
  const float* agent_qs = (const float*)d_in[0];
  const float* states   = (const float*)d_in[1];
  const int*   actions  = (const int*)d_in[2];
  const float* sel_W1   = (const float*)d_in[3];
  const float* sel_b1   = (const float*)d_in[4];
  const float* sel_W2   = (const float*)d_in[5];
  const float* key_W1   = (const float*)d_in[6];
  const float* key_b1   = (const float*)d_in[7];
  const float* key_W2   = (const float*)d_in[8];
  const float* wh_W1    = (const float*)d_in[9];
  const float* wh_b1    = (const float*)d_in[10];
  const float* wh_W2    = (const float*)d_in[11];
  const float* wh_b2    = (const float*)d_in[12];
  const float* v_W1     = (const float*)d_in[13];
  const float* v_b1     = (const float*)d_in[14];
  const float* v_W2     = (const float*)d_in[15];
  const float* v_b2     = (const float*)d_in[16];

  char* ws = (char*)d_ws;
  __bf16* sbf   = (__bf16*)(ws + OFF_SBF);
  __bf16* wcat  = (__bf16*)(ws + OFF_WCAT);
  float*  bcat  = (float*)(ws + OFF_BCAT);
  __bf16* y1    = (__bf16*)(ws + OFF_Y1);
  __bf16* kw1   = (__bf16*)(ws + OFF_KW1);
  float*  wqp   = (float*)(ws + OFF_WQ);
  __bf16* wm    = (__bf16*)(ws + OFF_WM);
  float*  selk  = (float*)(ws + OFF_SELK);
  float*  hq    = (float*)(ws + OFF_HQ);
  float*  whb   = (float*)(ws + OFF_WH);
  float*  vbuf  = (float*)(ws + OFF_V);
  float*  accum = (float*)(ws + OFF_ACC);
  float*  out   = (float*)d_out;

  // prep / conversion
  k_conv_states<<<(B_TOT * SP + 255) / 256, 256, 0, stream>>>(states, sbf);
  k_conv_w<<<(N1 * SP + 255) / 256, 256, 0, stream>>>(sel_W1, wh_W1, v_W1, sel_b1, wh_b1, v_b1,
                                                      wcat, bcat);
  k_conv_key<<<(NHEAD * KHID * 33 + 255) / 256, 256, 0, stream>>>(key_W1, kw1, wqp);
  k_combine<<<(NHEAD * KHID * HYP + 255) / 256, 256, 0, stream>>>(sel_W2, key_W2, wm);
  k_zero<<<1, 32, 0, stream>>>(accum);

  // big fused hyper layer-1 GEMM (WMMA + async double-buffer)
  k_gemm1<<<dim3(N1 / 128, B_TOT / 128), 256, 0, stream>>>(sbf, wcat, bcat, y1);
  // per-head selk GEMM (WMMA + async double-buffer)
  k_selk<<<dim3(NHEAD, B_TOT / 128), 256, 0, stream>>>(y1, wm, selk);
  // fused keys / logits / softmax / head_q / entropy / reg (WMMA + VALU)
  k_logits<<<B_TOT, 256, 0, stream>>>(sbf, agent_qs, actions, kw1, wqp, key_b1, selk, hq, accum);
  // wh / v heads
  k_whv<<<B_TOT / 8, 256, 0, stream>>>(y1, wh_W2, wh_b2, v_W2, v_b2, whb, vbuf);
  // final combine + scalars
  k_final<<<(B_TOT + 9 + 255) / 256, 256, 0, stream>>>(whb, hq, vbuf, accum, out);
}
